// MultiHeadAttention_16192026706153
// MI455X (gfx1250) — compile-verified, matrix operand order checked by reading, not run on hardware
//
#include <hip/hip_runtime.h>
#include <math.h>

// Problem constants (match reference)
#define H       8
#define DMODEL  512
#define DK      64
#define BATCH   4
#define SEQ     2048
#define MTOT    (BATCH * SEQ)                   // 8192 rows
#define OUT0    ((size_t)BATCH * SEQ * DMODEL)  // floats before attns region
#define INV_TEMPER 0.044194173824159216f        // 1/sqrt(512)

typedef __bf16        v16bf  __attribute__((ext_vector_type(16)));
typedef __bf16        bf16x4 __attribute__((ext_vector_type(4)));
typedef float         v8f    __attribute__((ext_vector_type(8)));
typedef unsigned int  u32x4  __attribute__((ext_vector_type(4)));
typedef float         f32x4  __attribute__((ext_vector_type(4)));
typedef int           i32x4  __attribute__((ext_vector_type(4)));
typedef int           i32x8  __attribute__((ext_vector_type(8)));

union Frag {
    v16bf  v;
    u32x4  q[2];
    bf16x4 b4[4];
    __bf16 u[16];
};

// ---------------------------------------------------------------------------
// Kernel 1: per-head Q/K/V projections.  X(8192x512 f32) @ W(512x64 f32)
// -> head tensor (8192x64 bf16).  grid = (64, 8 heads, 3 tensors), 256 thr.
// ---------------------------------------------------------------------------
__global__ __launch_bounds__(256)
void proj_kernel(const float* __restrict__ q, const float* __restrict__ k,
                 const float* __restrict__ v,
                 const float* __restrict__ wq, const float* __restrict__ wk,
                 const float* __restrict__ wv,
                 __bf16* __restrict__ qh, __bf16* __restrict__ kh,
                 __bf16* __restrict__ vh) {
    __shared__ __bf16 wt[64 * 32];   // B staged transposed: [n][kk] bf16

    const int tid  = threadIdx.x;
    const int wave = tid >> 5;
    const int lane = tid & 31;
    const int lane15 = lane & 15;
    const int koff   = (lane < 16) ? 0 : 8;

    const int head  = blockIdx.y;
    const int which = blockIdx.z;
    const float* X = (which == 0) ? q : (which == 1) ? k : v;
    const float* W = ((which == 0) ? wq : (which == 1) ? wk : wv) + (size_t)head * DMODEL * DK;
    __bf16* dst = ((which == 0) ? qh : (which == 1) ? kh : vh) + (size_t)head * MTOT * DK;

    const int m0   = blockIdx.x * 128 + wave * 16;
    const int arow = m0 + lane15;

    v8f acc[4] = {};

    for (int kc = 0; kc < 16; ++kc) {
        // Stage W chunk transposed: wt[n][kk] = bf16(W[kc*32+kk][n]).
        // Read contiguous along n (coalesced float4), scatter 2B into LDS.
        for (int e = tid; e < 512; e += 256) {
            int kk = e >> 4;             // 0..31
            int n4 = (e & 15) * 4;       // 0..60
            f32x4 f = *(const f32x4*)(W + (size_t)(kc * 32 + kk) * DK + n4);
            bf16x4 hv = __builtin_convertvector(f, bf16x4);
            wt[(n4 + 0) * 32 + kk] = hv.x;
            wt[(n4 + 1) * 32 + kk] = hv.y;
            wt[(n4 + 2) * 32 + kk] = hv.z;
            wt[(n4 + 3) * 32 + kk] = hv.w;
        }
        __syncthreads();

        // A fragment (16x32) direct from global fp32 -> packed bf16 converts
        Frag a;
        {
            const float* px = X + (size_t)arow * DMODEL + kc * 32 + koff;
            f32x4 f0 = *(const f32x4*)(px);
            f32x4 f1 = *(const f32x4*)(px + 4);
            f32x4 f2 = *(const f32x4*)(px + 16);
            f32x4 f3 = *(const f32x4*)(px + 20);
            a.b4[0] = __builtin_convertvector(f0, bf16x4);
            a.b4[1] = __builtin_convertvector(f1, bf16x4);
            a.b4[2] = __builtin_convertvector(f2, bf16x4);
            a.b4[3] = __builtin_convertvector(f3, bf16x4);
        }
#pragma unroll
        for (int t = 0; t < 4; ++t) {
            Frag b;
            const u32x4* pb = (const u32x4*)&wt[(t * 16 + lane15) * 32 + koff];
            b.q[0] = pb[0];
            b.q[1] = pb[2];   // +16 halfs = +32B
            acc[t] = __builtin_amdgcn_wmma_f32_16x16x32_bf16(
                false, a.v, false, b.v, (short)0, acc[t], false, false);
        }
        __syncthreads();
    }

    // Store bf16 head tensor
#pragma unroll
    for (int t = 0; t < 4; ++t) {
        int col = t * 16 + lane15;
#pragma unroll
        for (int r = 0; r < 8; ++r) {
            int row = m0 + r + ((lane < 16) ? 0 : 8);
            dst[(size_t)row * DK + col] = (__bf16)acc[t][r];
        }
    }
}

// ---------------------------------------------------------------------------
// Kernel 2: attention.  Per (h*B+b, 64-row tile): pass 1 = S = scale*QK^T via
// WMMA (K chunk staged by the Tensor Data Mover), raw S -> attns region,
// online row max/sum.  Pass 2 = p = exp(s-m)/sigma written back in place,
// P@V via WMMA.  grid = (32 row tiles, 32 hb), 128 threads (4 waves).
// ---------------------------------------------------------------------------
__global__ __launch_bounds__(128)
void attn_kernel(const __bf16* __restrict__ qh,
                 const __bf16* __restrict__ kh,
                 const __bf16* __restrict__ vh,
                 __bf16* __restrict__ concat,
                 float* __restrict__ S) {
    __shared__ __bf16 kbuf[64 * 64];   // 8 KB: K chunk [key][dim]
    __shared__ __bf16 vt[64 * 32];     // 4 KB: V chunk transposed [dv][key]
    __shared__ float stats[4][16][2];

    const int tid  = threadIdx.x;
    const int wave = tid >> 5;
    const int lane = tid & 31;
    const int lane15 = lane & 15;
    const int koff   = (lane < 16) ? 0 : 8;

    const int hb = blockIdx.y;          // h*B + b (matches attn.reshape order)
    const int h  = hb >> 2;
    const int b  = hb & 3;
    const int lq0 = blockIdx.x * 64 + wave * 16;

    const __bf16* qbase = qh + ((size_t)h * MTOT + (size_t)b * SEQ) * DK;
    const __bf16* kbase = kh + ((size_t)h * MTOT + (size_t)b * SEQ) * DK;
    const __bf16* vbase = vh + ((size_t)h * MTOT + (size_t)b * SEQ) * DK;
    float* Sbase = S + (size_t)hb * SEQ * SEQ;

    // ---- TDM descriptor (invariant parts): 1-D tile of 1024 x 8B = 8 KB ----
    // D# group1: data_size=3 (8B), tensor_dim0=1024, tensor_dim1=1,
    // tile_dim0=1024, tile_dim1/2=0 (unused), dim0_stride=1024.
    i32x8 g1;
    g1[0] = (3 << 16);                       // workgroup_mask=0, data_size=8B
    g1[1] = (int)((1024u & 0xFFFFu) << 16);  // tensor_dim0[15:0]
    g1[2] = (int)(1u << 16);                 // dim0[31:16]=0 | tensor_dim1[15:0]=1
    g1[3] = (int)(1024u << 16);              // dim1[31:16]=0 | tile_dim0=1024
    g1[4] = 0;                               // tile_dim1=0, tile_dim2=0
    g1[5] = 1024;                            // tensor_dim0_stride[31:0]
    g1[6] = 0;
    g1[7] = 0;
    const unsigned long long kaddr  = (unsigned long long)(size_t)kbase;
    const unsigned int       ldsoff = (unsigned int)(size_t)&kbuf[0];

    // Q A-fragments for both 32-wide k-steps (dims 0..31, 32..63)
    Frag aq[2];
#pragma unroll
    for (int ks = 0; ks < 2; ++ks) {
        const u32x4* p = (const u32x4*)(qbase + (size_t)(lq0 + lane15) * DK + ks * 32 + koff);
        aq[ks].q[0] = p[0];
        aq[ks].q[1] = p[2];
    }

    float rm[8], rs[8];
#pragma unroll
    for (int r = 0; r < 8; ++r) { rm[r] = -1e30f; rs[r] = 0.0f; }

    // -------- Pass 1: raw scaled scores + online stats --------
    for (int c = 0; c < 32; ++c) {
        // Stage K chunk via the Tensor Data Mover (one wave owns the DMA;
        // TDM ignores EXEC, done tracked with TENSORcnt).
        if (wave == 0) {
            unsigned long long ga = kaddr + (unsigned long long)c * (64u * DK * 2u);
            i32x4 g0;
            g0[0] = 1;                               // count=1 valid descriptor
            g0[1] = (int)ldsoff;                     // lds_addr (bytes)
            g0[2] = (int)(unsigned)(ga & 0xFFFFFFFFull);      // global_addr[31:0]
            g0[3] = (int)(((unsigned)(ga >> 32) & 0x01FFFFFFu)
                          | 0x80000000u);            // addr[56:32] | type=2
            asm volatile("tensor_load_to_lds %0, %1"
                         :: "s"(g0), "s"(g1) : "memory");
            __builtin_amdgcn_s_wait_tensorcnt(0);
        }
        __syncthreads();
#pragma unroll
        for (int t = 0; t < 4; ++t) {
            v8f acc = {};
#pragma unroll
            for (int ks = 0; ks < 2; ++ks) {
                Frag bk;
                const u32x4* p = (const u32x4*)&kbuf[(t * 16 + lane15) * DK + ks * 32 + koff];
                bk.q[0] = p[0];
                bk.q[1] = p[2];
                acc = __builtin_amdgcn_wmma_f32_16x16x32_bf16(
                    false, aq[ks].v, false, bk.v, (short)0, acc, false, false);
            }
#pragma unroll
            for (int r = 0; r < 8; ++r) {
                float s = acc[r] * INV_TEMPER;
                int lq = lq0 + r + ((lane < 16) ? 0 : 8);
                Sbase[(size_t)lq * SEQ + c * 64 + t * 16 + lane15] = s;
                // half-wave (16-lane) reductions: rows live in lane half-groups
                float vmax = s;
                for (int off = 8; off >= 1; off >>= 1)
                    vmax = fmaxf(vmax, __shfl_xor(vmax, off, 16));
                float mnew = fmaxf(rm[r], vmax);
                float e = __expf(s - mnew);
                float vsum = e;
                for (int off = 8; off >= 1; off >>= 1)
                    vsum += __shfl_xor(vsum, off, 16);
                rs[r] = rs[r] * __expf(rm[r] - mnew) + vsum;
                rm[r] = mnew;
            }
        }
        __syncthreads();
    }

    // Broadcast per-row stats through LDS so A-fragment lanes can read their row
    if (lane == 0 || lane == 16) {
        int rofs = (lane == 0) ? 0 : 8;
#pragma unroll
        for (int r = 0; r < 8; ++r) {
            stats[wave][rofs + r][0] = rm[r];
            stats[wave][rofs + r][1] = rs[r];
        }
    }
    __syncthreads();
    const float mrow = stats[wave][lane15][0];
    const float isig = 1.0f / stats[wave][lane15][1];

    // -------- Pass 2: normalize (write p in place) + P@V --------
    v8f oacc[4] = {};
    for (int c = 0; c < 64; ++c) {
        // stage V chunk transposed: vt[dv*32 + key] = vbase[(c*32+key)*64 + dv]
        for (int e = tid; e < 256; e += 128) {
            int key = e >> 3, part = e & 7;
            u32x4 raw = *(const u32x4*)(vbase + (size_t)(c * 32 + key) * DK + part * 8);
            const __bf16* rp = (const __bf16*)&raw;
#pragma unroll
            for (int j = 0; j < 8; ++j) vt[(part * 8 + j) * 32 + key] = rp[j];
        }
        if (c + 1 < 64) __builtin_prefetch(vbase + (size_t)(c + 1) * 32 * DK, 0, 1);
        __syncthreads();

        // P A-fragment: read raw s, softmax-normalize, write back, pack bf16
        Frag ap;
        float* srow = Sbase + (size_t)(lq0 + lane15) * SEQ + c * 32 + koff;
        f32x4 f0 = *(const f32x4*)(srow);
        f32x4 f1 = *(const f32x4*)(srow + 4);
        f32x4 f2 = *(const f32x4*)(srow + 16);
        f32x4 f3 = *(const f32x4*)(srow + 20);
#define SOFTMAX4(F) \
        F.x = __expf(F.x - mrow) * isig; F.y = __expf(F.y - mrow) * isig; \
        F.z = __expf(F.z - mrow) * isig; F.w = __expf(F.w - mrow) * isig;
        SOFTMAX4(f0) SOFTMAX4(f1) SOFTMAX4(f2) SOFTMAX4(f3)
#undef SOFTMAX4
        *(f32x4*)(srow)      = f0;
        *(f32x4*)(srow + 4)  = f1;
        *(f32x4*)(srow + 16) = f2;
        *(f32x4*)(srow + 20) = f3;
        ap.b4[0] = __builtin_convertvector(f0, bf16x4);
        ap.b4[1] = __builtin_convertvector(f1, bf16x4);
        ap.b4[2] = __builtin_convertvector(f2, bf16x4);
        ap.b4[3] = __builtin_convertvector(f3, bf16x4);

#pragma unroll
        for (int t = 0; t < 4; ++t) {
            Frag bv;
            const u32x4* p = (const u32x4*)&vt[(t * 16 + lane15) * 32 + koff];
            bv.q[0] = p[0];
            bv.q[1] = p[2];
            oacc[t] = __builtin_amdgcn_wmma_f32_16x16x32_bf16(
                false, ap.v, false, bv.v, (short)0, oacc[t], false, false);
        }
        __syncthreads();
    }

    // Store concat(b, lq, h*64+dv) as bf16 for the output projection
#pragma unroll
    for (int t = 0; t < 4; ++t) {
        int dv = t * 16 + lane15;
#pragma unroll
        for (int r = 0; r < 8; ++r) {
            int lq = lq0 + r + ((lane < 16) ? 0 : 8);
            concat[((size_t)(b * SEQ + lq)) * DMODEL + h * DK + dv] = (__bf16)oacc[t][r];
        }
    }
}

// ---------------------------------------------------------------------------
// Kernel 3: out = concat(8192x512 bf16) @ proj_w^T + proj_b + residual -> f32
// grid = (64 M-tiles of 128, 4 N-tiles of 128), 256 thr (8 waves x 16 rows).
// ---------------------------------------------------------------------------
__global__ __launch_bounds__(256)
void outproj_kernel(const __bf16* __restrict__ concat,
                    const float* __restrict__ projw,
                    const float* __restrict__ projb,
                    const float* __restrict__ resid,
                    float* __restrict__ zbuf) {
    __shared__ __bf16 wb[128 * 32];  // [n_local][kk] bf16 (8 KB)

    const int tid  = threadIdx.x;
    const int wave = tid >> 5;
    const int lane = tid & 31;
    const int lane15 = lane & 15;
    const int koff   = (lane < 16) ? 0 : 8;

    const int m0 = blockIdx.x * 128;
    const int n0 = blockIdx.y * 128;
    const int arow = m0 + wave * 16 + lane15;

    v8f acc[8] = {};

    for (int kc = 0; kc < 16; ++kc) {
        // Stage proj_w chunk (rows are output-n, contiguous in k -> no transpose)
        for (int e = tid; e < 1024; e += 256) {
            int nl = e >> 3, part = e & 7;
            f32x4 f = *(const f32x4*)(projw + (size_t)(n0 + nl) * DMODEL + kc * 32 + part * 4);
            *(bf16x4*)&wb[nl * 32 + part * 4] = __builtin_convertvector(f, bf16x4);
        }
        __syncthreads();

        Frag a;
        const u32x4* pa = (const u32x4*)(concat + (size_t)arow * DMODEL + kc * 32 + koff);
        a.q[0] = pa[0];
        a.q[1] = pa[2];
#pragma unroll
        for (int t = 0; t < 8; ++t) {
            Frag b;
            const u32x4* pb = (const u32x4*)&wb[(t * 16 + lane15) * 32 + koff];
            b.q[0] = pb[0];
            b.q[1] = pb[2];
            acc[t] = __builtin_amdgcn_wmma_f32_16x16x32_bf16(
                false, a.v, false, b.v, (short)0, acc[t], false, false);
        }
        __syncthreads();
    }

#pragma unroll
    for (int t = 0; t < 8; ++t) {
        int col = n0 + t * 16 + lane15;
        float bias = projb[col];
#pragma unroll
        for (int r = 0; r < 8; ++r) {
            int row = m0 + wave * 16 + r + ((lane < 16) ? 0 : 8);
            zbuf[(size_t)row * DMODEL + col] =
                acc[t][r] + bias + resid[(size_t)row * DMODEL + col];
        }
    }
}

// ---------------------------------------------------------------------------
// Kernel 4: LayerNorm with unbiased std (ddof=1) and (sigma+eps) denominator.
// One wave per row (512 elems -> 16/lane).  grid = 2048, 128 thr (4 rows/WG).
// ---------------------------------------------------------------------------
__global__ __launch_bounds__(128)
void ln_kernel(const float* __restrict__ z, const float* __restrict__ a2,
               const float* __restrict__ b2, float* __restrict__ out) {
    const int wave = threadIdx.x >> 5;
    const int lane = threadIdx.x & 31;
    const int row  = blockIdx.x * 4 + wave;
    const float* zr = z + (size_t)row * DMODEL;

    float s = 0.0f, ss = 0.0f;
    float vals[16];
#pragma unroll
    for (int i = 0; i < 16; ++i) {
        float x = zr[lane + 32 * i];
        vals[i] = x;
        s += x;
        ss += x * x;
    }
    for (int off = 16; off >= 1; off >>= 1) {
        s  += __shfl_xor(s, off, 32);
        ss += __shfl_xor(ss, off, 32);
    }
    float mu  = s * (1.0f / 512.0f);
    float var = (ss - 512.0f * mu * mu) * (1.0f / 511.0f);   // unbiased
    float inv = 1.0f / (sqrtf(fmaxf(var, 0.0f)) + 1e-3f);

    float* orow = out + (size_t)row * DMODEL;
#pragma unroll
    for (int i = 0; i < 16; ++i) {
        int c = lane + 32 * i;
        orow[c] = (vals[i] - mu) * inv * a2[c] + b2[c];
    }
}

// ---------------------------------------------------------------------------
extern "C" void kernel_launch(void* const* d_in, const int* in_sizes, int n_in,
                              void* d_out, int out_size, void* d_ws, size_t ws_size,
                              hipStream_t stream) {
    const float* q     = (const float*)d_in[0];
    const float* k     = (const float*)d_in[1];
    const float* v     = (const float*)d_in[2];
    const float* wq    = (const float*)d_in[3];
    const float* wk    = (const float*)d_in[4];
    const float* wv    = (const float*)d_in[5];
    const float* projw = (const float*)d_in[6];
    const float* projb = (const float*)d_in[7];
    const float* a2    = (const float*)d_in[8];
    const float* b2    = (const float*)d_in[9];

    float* out   = (float*)d_out;
    float* attns = out + OUT0;   // attns region of the concatenated output

    // Workspace layout (48 MB total, all offsets 1 MB aligned):
    char* ws = (char*)d_ws;
    __bf16* qh     = (__bf16*)(ws + 0);
    __bf16* kh     = (__bf16*)(ws + (size_t)8  * 1024 * 1024);
    __bf16* vh     = (__bf16*)(ws + (size_t)16 * 1024 * 1024);
    __bf16* concat = (__bf16*)(ws + (size_t)24 * 1024 * 1024);
    float*  zbuf   = (float*) (ws + (size_t)32 * 1024 * 1024);

    proj_kernel<<<dim3(64, H, 3), 256, 0, stream>>>(q, k, v, wq, wk, wv, qh, kh, vh);
    attn_kernel<<<dim3(SEQ / 64, H * BATCH), 128, 0, stream>>>(qh, kh, vh, concat, attns);
    outproj_kernel<<<dim3(MTOT / 128, DMODEL / 128), 256, 0, stream>>>(concat, projw, projb, q, zbuf);
    ln_kernel<<<dim3(MTOT / 4), 128, 0, stream>>>(zbuf, a2, b2, out);
}
